// FourierLayer_24000277250687
// MI455X (gfx1250) — compile-verified, matrix operand order checked by reading, not run on hardware
//
#include <hip/hip_runtime.h>
#include <hip/hip_bf16.h>

typedef __attribute__((ext_vector_type(16))) _Float16 v16h;
typedef __attribute__((ext_vector_type(8)))  _Float16 v8h;
typedef __attribute__((ext_vector_type(8)))  float    v8f;

#define NMODES 12
#define CIN   64
#define COUT  64
#define NB    8
#define NH    256
#define NW    256
#define TWO_PI 6.28318530717958647692f
#define INV_HW (1.0f/65536.0f)

// workspace byte offsets
#define OFF_F   0u            // 32x256 f16 forward DFT table (rows 24..31 = 0)
#define OFF_G   16384u        // 256x32 f16 inverse DFT table (cols 24..31 = 0)
#define OFF_T   32768u        // T[b][h][j24][c64] f32   (12.58 MB)
#define OFF_XF  12615680u     // Xf[b][m][n][ri][c] f32  (0.59 MB)
#define OFF_WT  13205504u     // Wt[ri][mn][c][o] f32    (4.72 MB)
#define OFF_OF  17924096u     // Of[b][o][mn][ri] f32    (0.59 MB)
#define OFF_U   18513920u     // U16[b][h][o][j32] f16   (8.39 MB)
#define WS_NEED 26902528u

union V8FU { v8f v; float f[8]; };

// ---------------- K0: twiddle tables ----------------
__global__ __launch_bounds__(256) void k_init_tables(_Float16* Ftab, _Float16* Gtab) {
    int idx = blockIdx.x * 256 + threadIdx.x;
    if (idx < 32 * 256) {                 // F[j][w], j<12: cos, 12..23: -sin, else 0
        int j = idx >> 8, w = idx & 255;
        float v = 0.0f;
        if (j < NMODES) v = __cosf((float)((j * w) & 255) * (TWO_PI / 256.0f));
        else if (j < 2 * NMODES) v = -__sinf((float)(((j - NMODES) * w) & 255) * (TWO_PI / 256.0f));
        Ftab[idx] = (_Float16)v;
    } else {                               // G[w][j], j<12: cos, 12..23: -sin, else 0
        int i2 = idx - 32 * 256;
        int w = i2 >> 5, j = i2 & 31;
        float v = 0.0f;
        if (j < NMODES) v = __cosf((float)((j * w) & 255) * (TWO_PI / 256.0f));
        else if (j < 2 * NMODES) v = -__sinf((float)(((j - NMODES) * w) & 255) * (TWO_PI / 256.0f));
        Gtab[i2] = (_Float16)v;
    }
}

// ---------------- K0b: transpose weights to [ri][mn][c][o] ----------------
__global__ __launch_bounds__(256) void k_transpose_w(const float* wr, const float* wi, float* Wt) {
    int tid = blockIdx.x * 256 + threadIdx.x;            // 2*144*4096 total
    int plane = tid / (144 * 4096);
    int i = tid % (144 * 4096);
    int mn = i >> 12;
    int co = i & 4095;
    const float* src = plane ? wi : wr;
    Wt[(size_t)plane * 144 * 4096 + i] = src[co * 144 + mn];   // w[c][o][m][n]
}

// ---------------- K1: forward DFT over w (WMMA) ----------------
// per (b,h): T(24x64) = F(24x256) * x_f16(256x64)
__global__ __launch_bounds__(256) void k_fwd_w(const float* __restrict__ x,
                                               const _Float16* __restrict__ Fg,
                                               float* __restrict__ Tw) {
    __shared__ _Float16 Xs[64 * 264];                     // [c][w], padded rows
    int b = blockIdx.x >> 8, h = blockIdx.x & 255;
    int t = threadIdx.x;

    // stage x[b,h,:,:] into LDS transposed, f32 -> f16
    const float4* xp = (const float4*)(x + ((size_t)(b * NH + h)) * NW * CIN);
#pragma unroll
    for (int i = 0; i < 16; ++i) {
        int id = i * 256 + t;                              // 4096 float4s
        int w = id >> 4, c0 = (id & 15) * 4;
        float4 v = xp[id];
        Xs[(c0 + 0) * 264 + w] = (_Float16)v.x;
        Xs[(c0 + 1) * 264 + w] = (_Float16)v.y;
        Xs[(c0 + 2) * 264 + w] = (_Float16)v.z;
        Xs[(c0 + 3) * 264 + w] = (_Float16)v.w;
    }
    __syncthreads();

    int wave = t >> 5, lane = t & 31;
    int m_tile = wave >> 2, n_tile = wave & 3;
    int row = lane & 15, hi = lane >> 4;
    int c0n = n_tile * 16;

    const _Float16* Fp = Fg + (m_tile * 16 + row) * 256 + hi * 8;
    const _Float16* Bp = &Xs[(c0n + row) * 264 + hi * 8];

    v8f acc = {};
#pragma unroll
    for (int kk = 0; kk < 8; ++kk) {
        int k0 = kk * 32;
        v8h aLo = *(const v8h*)(Fp + k0);
        v8h aHi = *(const v8h*)(Fp + k0 + 16);
        v16h a = __builtin_shufflevector(aLo, aHi, 0,1,2,3,4,5,6,7,8,9,10,11,12,13,14,15);
        v8h bLo = *(const v8h*)(Bp + k0);
        v8h bHi = *(const v8h*)(Bp + k0 + 16);
        v16h bb = __builtin_shufflevector(bLo, bHi, 0,1,2,3,4,5,6,7,8,9,10,11,12,13,14,15);
        acc = __builtin_amdgcn_wmma_f32_16x16x32_f16(false, a, false, bb, (short)0, acc, false, false);
    }

    // rows 24..31 (m_tile==1 && hi==1) are zero padding: single uniform guard,
    // all 8 stores unconditional inside -> no per-element exec juggling.
    if (m_tile == 0 || hi == 0) {
        float* Tp = Tw + ((size_t)(b * NH + h)) * 24 * CIN
                       + (size_t)(m_tile * 16 + hi * 8) * CIN + c0n + row;
        V8FU u; u.v = acc;
#pragma unroll
        for (int v = 0; v < 8; ++v) Tp[v * CIN] = u.f[v];
    }
}

// ---------------- K2: forward DFT over h ----------------
__global__ __launch_bounds__(64) void k_fwd_h(const float* __restrict__ Tw, float* __restrict__ Xf) {
    int b = blockIdx.x / NMODES, n = blockIdx.x % NMODES;
    int c = threadIdx.x;
    const float* Tb = Tw + (size_t)b * NH * 24 * CIN;
    float ar[NMODES], ai[NMODES];
#pragma unroll
    for (int m = 0; m < NMODES; ++m) { ar[m] = 0.f; ai[m] = 0.f; }
    for (int h = 0; h < NH; ++h) {
        float tre = Tb[(h * 24 + n) * CIN + c];
        float tim = Tb[(h * 24 + NMODES + n) * CIN + c];
#pragma unroll
        for (int m = 0; m < NMODES; ++m) {
            float s, co;
            __sincosf((float)((m * h) & 255) * (TWO_PI / 256.0f), &s, &co);
            ar[m] += tre * co + tim * s;      // (tre + i tim) * e^{-i a}
            ai[m] += tim * co - tre * s;
        }
    }
#pragma unroll
    for (int m = 0; m < NMODES; ++m) {
        float* dst = Xf + ((size_t)((b * NMODES + m) * NMODES + n)) * 128;
        dst[c] = ar[m];
        dst[64 + c] = ai[m];
    }
}

// ---------------- K3: complex channel einsum ----------------
__global__ __launch_bounds__(64) void k_einsum(const float* __restrict__ Xf,
                                               const float* __restrict__ Wt,
                                               float* __restrict__ Of) {
    int b = blockIdx.x / 144, mn = blockIdx.x % 144;
    int o = threadIdx.x;
    __shared__ float xr[64], xi[64];
    const float* xs = Xf + ((size_t)(b * 144 + mn)) * 128;
    xr[o] = xs[o];
    xi[o] = xs[64 + o];
    __syncthreads();
    const float* wrp = Wt + (size_t)mn * 4096;
    const float* wip = Wt + (size_t)144 * 4096 + (size_t)mn * 4096;
    float accr = 0.f, acci = 0.f;
#pragma unroll 4
    for (int c = 0; c < CIN; ++c) {
        float wrv = wrp[c * 64 + o], wiv = wip[c * 64 + o];
        accr += xr[c] * wrv - xi[c] * wiv;
        acci += xr[c] * wiv + xi[c] * wrv;
    }
    float* dst = Of + ((size_t)((b * COUT + o) * 144 + mn)) * 2;
    dst[0] = accr;
    dst[1] = acci;
}

// ---------------- K4: inverse DFT over m, emit f16 V rows ----------------
__global__ __launch_bounds__(256) void k_inv_h(const float* __restrict__ Of, _Float16* __restrict__ U) {
    __shared__ float of[32 * 144 * 2];                    // 36 KB
    int blk = blockIdx.x;
    int b = blk >> 3, rest = blk & 7;
    int hc = rest >> 1, oh = rest & 1;
    int t = threadIdx.x;
    const float* Ofb = Of + ((size_t)(b * COUT + oh * 32)) * 288;
    for (int i = t; i < 32 * 288; i += 256) of[i] = Ofb[i];
    __syncthreads();

    int o_l = t & 31, hh0 = t >> 5;
    int o = oh * 32 + o_l;
#pragma unroll
    for (int hs = 0; hs < 8; ++hs) {
        int h = hc * 64 + hh0 * 8 + hs;
        float cm[NMODES], sm[NMODES];
#pragma unroll
        for (int m = 0; m < NMODES; ++m)
            __sincosf((float)((m * h) & 255) * (TWO_PI / 256.0f), &sm[m], &cm[m]);
        union { v8h v[4]; _Float16 hh[32]; } ub;
#pragma unroll
        for (int n = 0; n < NMODES; ++n) {
            float ur = 0.f, ui = 0.f;
#pragma unroll
            for (int m = 0; m < NMODES; ++m) {
                const float* p = &of[(o_l * 144 + m * NMODES + n) * 2];
                float fr = p[0], fi = p[1];
                ur += fr * cm[m] - fi * sm[m];             // (fr + i fi) * e^{+i a}
                ui += fr * sm[m] + fi * cm[m];
            }
            ub.hh[n] = (_Float16)(ur * INV_HW);
            ub.hh[NMODES + n] = (_Float16)(ui * INV_HW);
        }
#pragma unroll
        for (int j = 24; j < 32; ++j) ub.hh[j] = (_Float16)0.0f;
        _Float16* dst = U + ((size_t)((b * NH + h) * COUT + o)) * 32;
        v8h* dv = (v8h*)dst;
        dv[0] = ub.v[0]; dv[1] = ub.v[1]; dv[2] = ub.v[2]; dv[3] = ub.v[3];
    }
}

// ---------------- K5: inverse DFT over w (WMMA) + final output ----------------
// per (b,h): out(256x64) = G(256x32) * V(32x64)
__global__ __launch_bounds__(256) void k_inv_w(const _Float16* __restrict__ U,
                                               const _Float16* __restrict__ Gg,
                                               float* __restrict__ out) {
    int b = blockIdx.x >> 8, h = blockIdx.x & 255;
    int t = threadIdx.x;
    int wave = t >> 5, lane = t & 31;
    int n_tile = wave & 3, half_m = wave >> 2;
    int row = lane & 15, hi = lane >> 4;

    // B fragment: V[j][o] with o = n_tile*16+row, same K mapping as A
    const _Float16* Ub = U + ((size_t)((b * NH + h) * COUT + n_tile * 16 + row)) * 32 + hi * 8;
    v8h bLo = *(const v8h*)(Ub);
    v8h bHi = *(const v8h*)(Ub + 16);
    v16h bb = __builtin_shufflevector(bLo, bHi, 0,1,2,3,4,5,6,7,8,9,10,11,12,13,14,15);

    float* outp = out + ((size_t)(b * NH + h)) * NW * COUT + n_tile * 16 + row;
#pragma unroll
    for (int mt = 0; mt < 8; ++mt) {
        int w0 = (half_m * 8 + mt) * 16;
        const _Float16* Gp = Gg + (w0 + row) * 32 + hi * 8;
        v8h aLo = *(const v8h*)(Gp);
        v8h aHi = *(const v8h*)(Gp + 16);
        v16h a = __builtin_shufflevector(aLo, aHi, 0,1,2,3,4,5,6,7,8,9,10,11,12,13,14,15);
        v8f acc = {};
        acc = __builtin_amdgcn_wmma_f32_16x16x32_f16(false, a, false, bb, (short)0, acc, false, false);
        V8FU u; u.v = acc;
        float* op = outp + (size_t)(w0 + hi * 8) * COUT;
#pragma unroll
        for (int v = 0; v < 8; ++v) op[(size_t)v * COUT] = u.f[v];
    }
}

extern "C" void kernel_launch(void* const* d_in, const int* in_sizes, int n_in,
                              void* d_out, int out_size, void* d_ws, size_t ws_size,
                              hipStream_t stream) {
    if (ws_size < WS_NEED) return;
    const float* x  = (const float*)d_in[0];
    const float* wr = (const float*)d_in[1];
    const float* wi = (const float*)d_in[2];
    float* out = (float*)d_out;
    char* ws = (char*)d_ws;

    _Float16* Ftab = (_Float16*)(ws + OFF_F);
    _Float16* Gtab = (_Float16*)(ws + OFF_G);
    float*    Tw   = (float*)   (ws + OFF_T);
    float*    Xf   = (float*)   (ws + OFF_XF);
    float*    Wt   = (float*)   (ws + OFF_WT);
    float*    Of   = (float*)   (ws + OFF_OF);
    _Float16* U    = (_Float16*)(ws + OFF_U);

    k_init_tables<<<64, 256, 0, stream>>>(Ftab, Gtab);
    k_transpose_w<<<4608, 256, 0, stream>>>(wr, wi, Wt);
    k_fwd_w<<<NB * NH, 256, 0, stream>>>(x, Ftab, Tw);
    k_fwd_h<<<NB * NMODES, 64, 0, stream>>>(Tw, Xf);
    k_einsum<<<NB * 144, 64, 0, stream>>>(Xf, Wt, Of);
    k_inv_h<<<64, 256, 0, stream>>>(Of, U);
    k_inv_w<<<NB * NH, 256, 0, stream>>>(U, Gtab, out);
}
